// Lookup_58849641890538
// MI455X (gfx1250) — compile-verified, matrix operand order checked by reading, not run on hardware
//
#include <hip/hip_runtime.h>

typedef __attribute__((ext_vector_type(16))) _Float16 v16h;
typedef __attribute__((ext_vector_type(8)))  _Float16 v8h;
typedef __attribute__((ext_vector_type(8)))  float    v8f;

#define B_   4
#define D_   256
#define H8_  48
#define W8_  64
#define NP_  3072      // H8*W8
#define L_   41
#define NQ_  192       // padded candidate count
#define NQV_ 187       // valid candidates: 10*12 + 6*7 + 4*4 + 3*3

// Diamond displacement tables (y in [-4,4], x in [|y|-4, 4-|y|])
__constant__ float c_YS[L_] = {
  -4,
  -3,-3,-3,
  -2,-2,-2,-2,-2,
  -1,-1,-1,-1,-1,-1,-1,
   0, 0, 0, 0, 0, 0, 0, 0, 0,
   1, 1, 1, 1, 1, 1, 1,
   2, 2, 2, 2, 2,
   3, 3, 3,
   4 };
__constant__ float c_XS[L_] = {
   0,
  -1, 0, 1,
  -2,-1, 0, 1, 2,
  -3,-2,-1, 0, 1, 2, 3,
  -4,-3,-2,-1, 0, 1, 2, 3, 4,
  -3,-2,-1, 0, 1, 2, 3,
  -2,-1, 0, 1, 2,
  -1, 0, 1,
   0 };

// ---------------------------------------------------------------------------
// Kernel 1: feat1 (B,D,H8,W8) f32  ->  A (b, p=i*64+j, d) f16, d contiguous.
// LDS-tiled 16x16 transpose for coalescing both sides.
// ---------------------------------------------------------------------------
__global__ __launch_bounds__(256) void k_feat1_pack(
    const float* __restrict__ f1, _Float16* __restrict__ A) {
  __shared__ float t[16][17];
  int blk = blockIdx.x;                 // B * (3072/16) * (256/16) blocks
  int dt  = blk & 15;                   // d tile
  int pt  = (blk >> 4) % 192;           // p tile
  int b   = (blk >> 4) / 192;
  int tx = threadIdx.x & 15, ty = threadIdx.x >> 4;
  // read: consecutive tx -> consecutive p (coalesced)
  t[ty][tx] = f1[((size_t)(b * D_ + dt * 16 + ty)) * NP_ + pt * 16 + tx];
  __syncthreads();
  // write: consecutive tx -> consecutive d (coalesced)
  A[((size_t)(b * NP_ + pt * 16 + ty)) * D_ + dt * 16 + tx] = (_Float16)t[tx][ty];
}

// ---------------------------------------------------------------------------
// Kernel 2: gather+avg-pool the candidate feat2 vectors.
// q layout: L0: 10x12 box (q=0..119), L1: 6x7 (120..161), L2: 4x4 (162..177),
//           L3: 3x3 (178..186); 187..191 zero padding.
// ---------------------------------------------------------------------------
__global__ __launch_bounds__(256) void k_build_cands(
    const float* __restrict__ f2, _Float16* __restrict__ Bm) {
  int blk = blockIdx.x;                 // B*NQ_ blocks
  int b = blk / NQ_;
  int q = blk % NQ_;
  int d = threadIdx.x;                  // 256 threads = one vector
  float val = 0.f;
  if (q < NQV_) {
    int level, y, x;
    if (q < 120)      { level = 0; y =  q        / 12; x =  q        % 12; }
    else if (q < 162) { level = 1; y = (q - 120) /  7; x = (q - 120) %  7; }
    else if (q < 178) { level = 2; y = (q - 162) /  4; x = (q - 162) %  4; }
    else              { level = 3; y = (q - 178) /  3; x = (q - 178) %  3; }
    int s = 1 << level;
    const float* base = f2 + ((size_t)(b * D_ + d)) * NP_;
    float sum = 0.f;
    for (int yy = 0; yy < s; ++yy)
      for (int xx = 0; xx < s; ++xx)
        sum += base[(y * s + yy) * W8_ + (x * s + xx)];
    val = sum / (float)(s * s);
  }
  Bm[((size_t)(b * NQ_ + q)) * D_ + d] = (_Float16)val;
}

// ---------------------------------------------------------------------------
// Kernel 3: main. One block = (b, row i, 16-pixel tile j0..j0+15).
// Phase 1: WMMA GEMM 16 pixels x 192 candidates, K=256 -> LDS slab.
// Phase 2: bilinear lookups from the LDS slab -> output (B,41,4,48,64).
// ---------------------------------------------------------------------------
__global__ __launch_bounds__(128) void k_corr_lookup(
    const _Float16* __restrict__ A, const _Float16* __restrict__ Bm,
    const float* __restrict__ flow, float* __restrict__ out) {
  __shared__ float Cst[16 * 193];       // padded stride (193%64!=0): no bank conflict
  __shared__ float sfx[16], sfy[16];

  int blk = blockIdx.x;                 // B*48*4 = 768
  int b   = blk / 192;
  int rem = blk % 192;
  int i   = rem >> 2;
  int j0  = (rem & 3) << 4;

  int tid  = threadIdx.x;
  int lane = tid & 31;
  int wave = tid >> 5;
  int m    = lane & 15;                 // M row (A) / N col (B)
  int hi   = lane >> 4;                 // K-half selector per ISA layout

  if (tid < 16) {
    sfy[tid] = flow[          i * W8_ + j0 + tid];   // curr_flow[0] = fy
    sfx[tid] = flow[NP_ +     i * W8_ + j0 + tid];   // curr_flow[1] = fx
  }

  // Preload all 8 A fragments (K=0..255) for this pixel tile.
  // 16-bit A 16x32 layout: lane(0-15) halves0-7 = K hi*8+0..7, halves8-15 = K 16+hi*8+0..7
  const _Float16* arow = A + ((size_t)(b * NP_ + i * W8_ + j0 + m)) * D_;
  v16h af[8];
#pragma unroll
  for (int kk = 0; kk < 8; ++kk) {
    v8h lo = *(const v8h*)(arow + kk * 32 + hi * 8);
    v8h hh = *(const v8h*)(arow + kk * 32 + 16 + hi * 8);
    v16h a;
#pragma unroll
    for (int t = 0; t < 8; ++t) { a[t] = lo[t]; a[t + 8] = hh[t]; }
    af[kk] = a;
  }

  // Each wave owns 3 of the 12 N-tiles (uniform per wave -> EXEC all ones at WMMA).
#pragma unroll
  for (int qi = 0; qi < 3; ++qi) {
    int qt = wave * 3 + qi;
    const _Float16* brow = Bm + ((size_t)(b * NQ_ + qt * 16 + m)) * D_;
    v8f c = {0.f, 0.f, 0.f, 0.f, 0.f, 0.f, 0.f, 0.f};
#pragma unroll
    for (int kk = 0; kk < 8; ++kk) {
      v8h lo = *(const v8h*)(brow + kk * 32 + hi * 8);
      v8h hh = *(const v8h*)(brow + kk * 32 + 16 + hi * 8);
      v16h bf;
#pragma unroll
      for (int t = 0; t < 8; ++t) { bf[t] = lo[t]; bf[t + 8] = hh[t]; }
      c = __builtin_amdgcn_wmma_f32_16x16x32_f16(
              /*neg_a=*/false, af[kk], /*neg_b=*/false, bf,
              /*c_mod=*/(short)0, c, /*reuse_a=*/false, /*reuse_b=*/false);
    }
    // C layout: VGPR r -> (M = r + 8*hi, N = lane&15)
#pragma unroll
    for (int r = 0; r < 8; ++r)
      Cst[(r + 8 * hi) * 193 + qt * 16 + m] = c[r];
  }
  __syncthreads();

  // Phase 2: bilinear resolve. px = ((j+fx)/2^k + xs)*(wk-1)*2^k/W  (align_corners).
  const int   qb_[4] = {0, 120, 162, 178};
  const int   bW_[4] = {12, 7, 4, 3};
  const int   bH_[4] = {10, 6, 4, 3};
  const int   wk_[4] = {64, 32, 16, 8};
  const int   hk_[4] = {48, 24, 12, 6};
  const float cx_[4] = {63.f/512.f, 62.f/512.f, 60.f/512.f, 56.f/512.f};
  const float cy_[4] = {47.f/384.f, 46.f/384.f, 44.f/384.f, 40.f/384.f};
  const float is_[4] = {1.f, 0.5f, 0.25f, 0.125f};

  for (int e = tid; e < 16 * 4 * L_; e += 128) {
    int mm    = e & 15;
    int r     = e >> 4;       // 0..163
    int level = r & 3;
    int l     = r >> 2;       // 0..40
    float fx = sfx[mm], fy = sfy[mm];
    float px = (((float)(j0 + mm) + fx) * is_[level] + c_XS[l]) * cx_[level];
    float py = (((float)i        + fy) * is_[level] + c_YS[l]) * cy_[level];
    float fx0 = floorf(px), fy0 = floorf(py);
    int   x0 = (int)fx0,    y0 = (int)fy0;
    float tx = px - fx0,    ty = py - fy0;
    float acc = 0.f;
#pragma unroll
    for (int dy = 0; dy < 2; ++dy)
#pragma unroll
      for (int dx = 0; dx < 2; ++dx) {
        int xi = x0 + dx, yi = y0 + dy;
        float w = (dx ? tx : 1.f - tx) * (dy ? ty : 1.f - ty);
        if (xi >= 0 && yi >= 0 && xi < wk_[level] && yi < hk_[level] &&
            xi < bW_[level] && yi < bH_[level])
          acc += w * Cst[mm * 193 + qb_[level] + yi * bW_[level] + xi];
      }
    out[((((size_t)b * L_ + l) * 4 + level) * NP_) + i * W8_ + j0 + mm] = acc * 0.0625f;
  }
}

// ---------------------------------------------------------------------------
extern "C" void kernel_launch(void* const* d_in, const int* in_sizes, int n_in,
                              void* d_out, int out_size, void* d_ws, size_t ws_size,
                              hipStream_t stream) {
  (void)in_sizes; (void)n_in; (void)out_size; (void)ws_size;
  const float* feat1 = (const float*)d_in[0];   // (4,256,48,64)
  const float* feat2 = (const float*)d_in[1];   // (4,256,48,64)
  const float* flow  = (const float*)d_in[2];   // (2,48,64)
  float* out = (float*)d_out;                   // (4,41,4,48,64)

  _Float16* Amat = (_Float16*)d_ws;                           // 4*3072*256*2 = 6291456 B
  _Float16* Bm   = (_Float16*)((char*)d_ws + 6291456);        // 4*192*256*2  =  393216 B

  k_feat1_pack <<<B_ * 192 * 16, 256, 0, stream>>>(feat1, Amat);
  k_build_cands<<<B_ * NQ_,      256, 0, stream>>>(feat2, Bm);
  k_corr_lookup<<<B_ * 48 * 4,   128, 0, stream>>>(Amat, Bm, flow, out);
}
